// WaveNet_69337952026871
// MI455X (gfx1250) — compile-verified, matrix operand order checked by reading, not run on hardware
//
#include <hip/hip_runtime.h>
#include <hip/hip_bf16.h>

typedef __attribute__((ext_vector_type(16))) __bf16 v16bf;
typedef __attribute__((ext_vector_type(8)))  __bf16 v8bf;
typedef __attribute__((ext_vector_type(8)))  float  v8f;
typedef __attribute__((ext_vector_type(4)))  int    v4i;

#define T_LEN 8192

// ---- gfx1250 async global->LDS path (guarded; falls back to sync copy) ----
#if defined(__has_builtin)
#  if __has_builtin(__builtin_amdgcn_global_load_async_to_lds_b128)
#    define WN_HAVE_ASYNC 1
#  endif
#  if __has_builtin(__builtin_amdgcn_s_wait_asynccnt)
#    define WN_HAVE_ASYNCCNT 1
#  endif
#endif

__device__ __forceinline__ void copy16_g2l(unsigned short* dst_lds, const unsigned short* src_glob) {
#if defined(WN_HAVE_ASYNC)
    typedef __attribute__((address_space(1))) v4i gv4i;   // global int4
    typedef __attribute__((address_space(3))) v4i lv4i;   // LDS int4
    __builtin_amdgcn_global_load_async_to_lds_b128((gv4i*)src_glob, (lv4i*)dst_lds, 0, 0);
#else
    *(uint4*)dst_lds = *(const uint4*)src_glob;
#endif
}
__device__ __forceinline__ void async_join() {
#if defined(WN_HAVE_ASYNC) && defined(WN_HAVE_ASYNCCNT)
    __builtin_amdgcn_s_wait_asynccnt(0);
#endif
}

__device__ __forceinline__ unsigned short f32_bf16(float f) {
    unsigned int u = __float_as_uint(f);
    u += 0x7FFFu + ((u >> 16) & 1u);          // round-to-nearest-even
    return (unsigned short)(u >> 16);
}
__device__ __forceinline__ float sigmoidf_(float x) { return 1.0f / (1.0f + __expf(-x)); }

// Load one 16x32 bf16 fragment from an LDS tile stored row-major [rows][32].
// CDNA5 16-bit A/B layout: lane<16 -> row=lane, K=0..7 & 16..23 ; lane>=16 -> K=8..15 & 24..31.
__device__ __forceinline__ v16bf load_frag(const unsigned short* lds_tile, int row0) {
    const int lane = threadIdx.x & 31;
    const int r  = row0 + (lane & 15);
    const int ko = (lane >> 4) * 8;
    v8bf lo = *(const v8bf*)(lds_tile + r * 32 + ko);
    v8bf hi = *(const v8bf*)(lds_tile + r * 32 + ko + 16);
    v16bf a;
#pragma unroll
    for (int i = 0; i < 8; i++) { a[i] = lo[i]; a[i + 8] = hi[i]; }
    return a;
}

// ---------------- prep kernels ----------------

// Pack per-layer A matrix: Abig[l][oc][1408] = [inW k=0 | k=1 | k=2 | condW slice] (bf16),
// Ars[l][oc][256] = rsW (bf16), bias1[l*512+oc] = in_b + cond_b.
__global__ void wn_prep_w(const float* __restrict__ in_w, const float* __restrict__ cond_w,
                          const float* __restrict__ rs_w, const float* __restrict__ in_b,
                          const float* __restrict__ cond_b,
                          unsigned short* __restrict__ Abig, unsigned short* __restrict__ Ars,
                          float* __restrict__ bias1) {
    int r = blockIdx.x * blockDim.x + threadIdx.x;        // 0..4095 == l*512+oc
    if (r >= 4096) return;
    unsigned short* dst = Abig + (size_t)r * 1408;
    const float* wsrc = in_w + (size_t)r * 768;           // [ic][k], k stride 1
    for (int ic = 0; ic < 256; ic++) {
#pragma unroll
        for (int k = 0; k < 3; k++) dst[k * 256 + ic] = f32_bf16(wsrc[ic * 3 + k]);
    }
    const float* csrc = cond_w + (size_t)r * 640;
    for (int m = 0; m < 640; m++) dst[768 + m] = f32_bf16(csrc[m]);
    const float* rsrc = rs_w + (size_t)r * 256;
    unsigned short* rdst = Ars + (size_t)r * 256;
    for (int ic = 0; ic < 256; ic++) rdst[ic] = f32_bf16(rsrc[ic]);
    bias1[r] = in_b[r] + cond_b[r];
}

__global__ void wn_prep_spect(const float* __restrict__ s, unsigned short* __restrict__ d, int n) {
    int i = (blockIdx.x * blockDim.x + threadIdx.x) * 4;
    if (i >= n) return;
    float4 v = *(const float4*)(s + i);
    d[i + 0] = f32_bf16(v.x); d[i + 1] = f32_bf16(v.y);
    d[i + 2] = f32_bf16(v.z); d[i + 3] = f32_bf16(v.w);
}

// start conv (4 in-ch 1x1) + zero the skip accumulator
__global__ void wn_start(const float* __restrict__ audio_in, const float* __restrict__ start_w,
                         const float* __restrict__ start_b,
                         float* __restrict__ audio_f32, unsigned short* __restrict__ audio_bf,
                         float* __restrict__ out_f32) {
    int idx = blockIdx.x * 256 + threadIdx.x;             // [b][c][t] over 4*256*8192
    int t = idx & 8191;
    int c = (idx >> 13) & 255;
    int b = idx >> 21;
    const float* ain = audio_in + (((size_t)b * 4) << 13) + t;
    float acc = start_b[c];
#pragma unroll
    for (int i = 0; i < 4; i++) acc += start_w[c * 4 + i] * ain[(size_t)i << 13];
    audio_f32[idx] = acc;
    audio_bf[idx]  = f32_bf16(acc);
    out_f32[idx]   = 0.0f;
}

// end conv (8 out-ch 1x1)
__global__ void wn_end(const float* __restrict__ out_f32, const float* __restrict__ end_w,
                       const float* __restrict__ end_b, float* __restrict__ d_out) {
    int idx = blockIdx.x * 256 + threadIdx.x;             // [b][oc][t] over 4*8*8192
    int t  = idx & 8191;
    int oc = (idx >> 13) & 7;
    int b  = idx >> 16;
    const float* src = out_f32 + (((size_t)b * 256) << 13) + t;
    float acc = end_b[oc];
    for (int ic = 0; ic < 256; ic++) acc += end_w[oc * 256 + ic] * src[(size_t)ic << 13];
    d_out[idx] = acc;
}

// ---------------- fused layer kernels ----------------

// Dilated in-conv + cond conv + bias + gated activation.
// Block: M=256 (128 tanh rows + 128 sigmoid rows), N=64 time cols, K=1408 in steps of 32.
// Double-buffered LDS pipeline: stage(next) overlaps WMMA(cur); one barrier per K-step.
__global__ __launch_bounds__(256) void wn_layer_gate(
        const unsigned short* __restrict__ Abig, const float* __restrict__ bias1,
        const unsigned short* __restrict__ audio_bf, const unsigned short* __restrict__ spect_bf,
        unsigned short* __restrict__ acts, int layer, int dil) {
    __shared__ __align__(16) unsigned short As[2][256 * 32];
    __shared__ __align__(16) unsigned short Bs[2][64 * 32];

    const int tid  = threadIdx.x;
    const int lane = tid & 31, wave = tid >> 5;
    const int t0 = blockIdx.x * 64;
    const int c0 = blockIdx.y * 128;                       // gate-channel tile
    const int b  = blockIdx.z;

    // A staging: this thread owns one LDS row; rows 0..127 tanh half, 128..255 sigmoid half
    const int arow_g = c0 + tid + ((tid & 128) ? 128 : 0); // Y row in [0,512)
    const unsigned short* arow_p = Abig + ((size_t)layer * 512 + arow_g) * 1408;
    // B staging: thread loads 8 time-contiguous elems at k-row bk, transposes into LDS
    const int bk = tid >> 3;
    const int bn = (tid & 7) * 8;

    auto stage = [&](int k0, int buf) {
        // A: 256 rows x 32 k of contiguous bf16 weights (async DMA to LDS when available)
        unsigned short* adst = &As[buf][tid * 32];
        const unsigned short* asrc = arow_p + k0;
#pragma unroll
        for (int q = 0; q < 4; q++) copy16_g2l(adst + q * 8, asrc + q * 8);
#ifndef WN_HAVE_ASYNC
        if (k0 + 32 < 1408) __builtin_prefetch(arow_p + k0 + 32, 0, 1);
#endif
        // B transposed: Bs[n][k]; first 768 K = shifted audio, rest = spect
        unsigned short vals[8];
        if (k0 < 768) {
            const int shift = dil * ((k0 >> 8) - 1);
            const int ic    = (k0 & 255) + bk;
            const unsigned short* row = audio_bf + (((size_t)b * 256 + ic) << 13);
#pragma unroll
            for (int j = 0; j < 8; j++) {
                int t = t0 + bn + j + shift;
                vals[j] = (t >= 0 && t < T_LEN) ? row[t] : (unsigned short)0;
            }
        } else {
            const unsigned short* row =
                spect_bf + (((size_t)b * 640 + (k0 - 768) + bk) << 13) + t0 + bn;
#pragma unroll
            for (int j = 0; j < 8; j++) vals[j] = row[j];
        }
#pragma unroll
        for (int j = 0; j < 8; j++) Bs[buf][(bn + j) * 32 + bk] = vals[j];
    };

    v8f acc_t[4] = {};
    v8f acc_s[4] = {};

    stage(0, 0);
    async_join();
    __syncthreads();

    const int NK = 1408 / 32;
    for (int kk = 0; kk < NK; kk++) {
        const int cur = kk & 1;
        if (kk + 1 < NK) stage((kk + 1) * 32, cur ^ 1);

        v16bf af_t = load_frag(As[cur], wave * 16);        // tanh rows  c0+16w..+15
        v16bf af_s = load_frag(As[cur], 128 + wave * 16);  // sigmoid rows (same channels +256)
#pragma unroll
        for (int nt = 0; nt < 4; nt++) {
            v16bf bf = load_frag(Bs[cur], nt * 16);
            acc_t[nt] = __builtin_amdgcn_wmma_f32_16x16x32_bf16(
                false, af_t, false, bf, (short)0, acc_t[nt], false, false);
            acc_s[nt] = __builtin_amdgcn_wmma_f32_16x16x32_bf16(
                false, af_s, false, bf, (short)0, acc_s[nt], false, false);
        }
        async_join();
        __syncthreads();
    }

    // gated activation fully in-register (tanh/sigmoid halves live in the same wave)
    const int hi = lane >> 4, ncol = lane & 15;
#pragma unroll
    for (int nt = 0; nt < 4; nt++) {
#pragma unroll
        for (int j = 0; j < 8; j++) {
            int row = c0 + wave * 16 + j + 8 * hi;     // channel in [0,256)
            float x = acc_t[nt][j] + bias1[layer * 512 + row];
            float y = acc_s[nt][j] + bias1[layer * 512 + 256 + row];
            float a = tanhf(x) * sigmoidf_(y);
            acts[(((size_t)b * 256 + row) << 13) + t0 + nt * 16 + ncol] = f32_bf16(a);
        }
    }
}

// Res/skip 1x1 conv (K=256) + residual/skip accumulation. Same double-buffered pipeline.
__global__ __launch_bounds__(256) void wn_layer_res(
        const unsigned short* __restrict__ Ars, const float* __restrict__ rs_b,
        const unsigned short* __restrict__ acts,
        float* __restrict__ audio_f32, unsigned short* __restrict__ audio_bf,
        float* __restrict__ out_f32, int layer, int last) {
    __shared__ __align__(16) unsigned short As[2][256 * 32];
    __shared__ __align__(16) unsigned short Bs[2][64 * 32];

    const int tid  = threadIdx.x;
    const int lane = tid & 31, wave = tid >> 5;
    const int t0 = blockIdx.x * 64;
    const int b  = blockIdx.z;
    const int m0     = last ? 0 : (int)blockIdx.y * 256;   // rows [m0,m0+256) of rs output
    const int to_out = last ? 1 : (blockIdx.y == 1);

    const unsigned short* arow_p = Ars + ((size_t)(layer * 512 + m0 + tid)) * 256;
    const int bk = tid >> 3;
    const int bn = (tid & 7) * 8;

    auto stage = [&](int k0, int buf) {
        unsigned short* adst = &As[buf][tid * 32];
        const unsigned short* asrc = arow_p + k0;
#pragma unroll
        for (int q = 0; q < 4; q++) copy16_g2l(adst + q * 8, asrc + q * 8);
        const unsigned short* row = acts + (((size_t)b * 256 + k0 + bk) << 13) + t0 + bn;
        unsigned short vals[8];
#pragma unroll
        for (int j = 0; j < 8; j++) vals[j] = row[j];
#pragma unroll
        for (int j = 0; j < 8; j++) Bs[buf][(bn + j) * 32 + bk] = vals[j];
    };

    v8f acc[2][4] = {};

    stage(0, 0);
    async_join();
    __syncthreads();

    const int NK = 256 / 32;
    for (int kk = 0; kk < NK; kk++) {
        const int cur = kk & 1;
        if (kk + 1 < NK) stage((kk + 1) * 32, cur ^ 1);

        v16bf a0 = load_frag(As[cur], wave * 32);
        v16bf a1 = load_frag(As[cur], wave * 32 + 16);
#pragma unroll
        for (int nt = 0; nt < 4; nt++) {
            v16bf bf = load_frag(Bs[cur], nt * 16);
            acc[0][nt] = __builtin_amdgcn_wmma_f32_16x16x32_bf16(
                false, a0, false, bf, (short)0, acc[0][nt], false, false);
            acc[1][nt] = __builtin_amdgcn_wmma_f32_16x16x32_bf16(
                false, a1, false, bf, (short)0, acc[1][nt], false, false);
        }
        async_join();
        __syncthreads();
    }

    const int hi = lane >> 4, ncol = lane & 15;
#pragma unroll
    for (int mt = 0; mt < 2; mt++) {
#pragma unroll
        for (int nt = 0; nt < 4; nt++) {
#pragma unroll
            for (int j = 0; j < 8; j++) {
                int rloc = wave * 32 + mt * 16 + j + 8 * hi;   // 0..255
                float v = acc[mt][nt][j] + rs_b[layer * 512 + m0 + rloc];
                size_t idx = (((size_t)b * 256 + rloc) << 13) + t0 + nt * 16 + ncol;
                if (to_out) {
                    out_f32[idx] += v;
                } else {
                    float na = audio_f32[idx] + v;
                    audio_f32[idx] = na;
                    audio_bf[idx]  = f32_bf16(na);
                }
            }
        }
    }
}

// ---------------- host launcher ----------------

extern "C" void kernel_launch(void* const* d_in, const int* in_sizes, int n_in,
                              void* d_out, int out_size, void* d_ws, size_t ws_size,
                              hipStream_t stream) {
    const float* spect   = (const float*)d_in[0];
    const float* audio   = (const float*)d_in[1];
    const float* start_w = (const float*)d_in[2];
    const float* start_b = (const float*)d_in[3];
    const float* cond_w  = (const float*)d_in[4];
    const float* cond_b  = (const float*)d_in[5];
    const float* in_w    = (const float*)d_in[6];
    const float* in_b    = (const float*)d_in[7];
    const float* rs_w    = (const float*)d_in[8];
    const float* rs_b    = (const float*)d_in[9];
    const float* end_w   = (const float*)d_in[10];
    const float* end_b   = (const float*)d_in[11];

    char* ws = (char*)d_ws;
    float*          audio_f32 = (float*)(ws + 0);                    // 4*256*8192 f32
    float*          out_f32   = (float*)(ws + 33554432);             // 4*256*8192 f32
    unsigned short* audio_bf  = (unsigned short*)(ws + 67108864);    // bf16 shadow
    unsigned short* acts      = (unsigned short*)(ws + 83886080);    // 4*256*8192 bf16
    unsigned short* spect_bf  = (unsigned short*)(ws + 100663296);   // 4*640*8192 bf16
    unsigned short* Abig      = (unsigned short*)(ws + 142606336);   // 8*512*1408 bf16
    unsigned short* Ars       = (unsigned short*)(ws + 154140672);   // 8*512*256 bf16
    float*          bias1     = (float*)(ws + 156237824);            // 8*512 f32

    wn_prep_w<<<16, 256, 0, stream>>>(in_w, cond_w, rs_w, in_b, cond_b, Abig, Ars, bias1);
    wn_prep_spect<<<20480, 256, 0, stream>>>(spect, spect_bf, 4 * 640 * 8192);
    wn_start<<<32768, 256, 0, stream>>>(audio, start_w, start_b, audio_f32, audio_bf, out_f32);

    for (int l = 0; l < 8; l++) {
        int dil  = 1 << l;
        int last = (l == 7);
        wn_layer_gate<<<dim3(T_LEN / 64, 2, 4), 256, 0, stream>>>(
            Abig, bias1, audio_bf, spect_bf, acts, l, dil);
        wn_layer_res<<<dim3(T_LEN / 64, last ? 1 : 2, 4), 256, 0, stream>>>(
            Ars, rs_b, acts, audio_f32, audio_bf, out_f32, l, last);
    }

    wn_end<<<1024, 256, 0, stream>>>(out_f32, end_w, end_b, (float*)d_out);
    (void)in_sizes; (void)n_in; (void)out_size; (void)ws_size;
}
